// AttentionBlock_3599182594677
// MI455X (gfx1250) — matrix operand order repaired and checked by reading, not compiled, not run on hardware
//
#include <hip/hip_runtime.h>
#include <hip/hip_bf16.h>

// ---------------------------------------------------------------------------
// Attention block (flash-attention style), bf16 WMMA on gfx1250 (CDNA5).
// B=4, H=W=64 -> N=4096 rows per batch, C=256, D=32.
// V stored transposed in ws so K/V tile staging is a straight contiguous
// copy -> GLOBAL_LOAD_ASYNC_TO_LDS_B128 double-buffered pipeline (ASYNCcnt).
// ---------------------------------------------------------------------------

typedef __bf16 bf16_t;
typedef __attribute__((ext_vector_type(16))) __bf16 v16bf;
typedef __attribute__((ext_vector_type(8)))  float  v8f;
typedef int v4i __attribute__((vector_size(4 * sizeof(int))));

union BFrag { v16bf v; bf16_t h[16]; unsigned u[8]; };
union FAcc  { v8f  v; float  f[8]; };

#define NROWS   16384      // B*N
#define CDIM    256
#define DDIM    32
#define NKEYS   4096       // per batch
#define NTILES  (NKEYS / 32)

#if __has_builtin(__builtin_amdgcn_global_load_async_to_lds_b128) && \
    __has_builtin(__builtin_amdgcn_s_wait_asynccnt)
#define USE_ASYNC_LDS 1
#else
#define USE_ASYNC_LDS 0
#endif

// Builtin signature (from clang diagnostic): param0 = v4i addrspace(1)*,
// param1 = v4i addrspace(3)*, then two immediate ints (offset, cpol).
#define GAS1(p) ((__attribute__((address_space(1))) v4i*)(p))
#define LAS3(p) ((__attribute__((address_space(3))) v4i*)(p))

// ---------------------------------------------------------------------------
// Kernel 1: convert + transpose weights to bf16 so WMMA B-fragments are
// contiguous 32-byte loads (K-pairs contiguous per lane, ISA 7.12.2).
// ---------------------------------------------------------------------------
__global__ void prep_weights(const float* __restrict__ Wq,
                             const float* __restrict__ Wk,
                             const float* __restrict__ Wv,
                             bf16_t* __restrict__ WtQ,
                             bf16_t* __restrict__ WtK,
                             bf16_t* __restrict__ WtV) {
  int i = blockIdx.x * blockDim.x + threadIdx.x;
  if (i < 32 * 256) {
    int c = i / 256, k = i % 256;
    WtQ[i] = (bf16_t)Wq[k * 32 + c];
  } else if (i < 2 * 32 * 256) {
    int j = i - 32 * 256;
    int c = j / 256, k = j % 256;
    WtK[j] = (bf16_t)Wk[k * 32 + c];
  } else if (i < 2 * 32 * 256 + 256 * 256) {
    int l = i - 2 * 32 * 256;
    int c = l / 256, k = l % 256;
    WtV[l] = (bf16_t)Wv[k * 256 + c];
  }
}

// ---------------------------------------------------------------------------
// Kernel 2: fused QKV projection. 16 rows/wave, 8 cached A fragments of x
// reused across 20 column tiles.  q,k stored row-major [n][d]; v stored
// TRANSPOSED [batch][c][n] so attention tile staging is contiguous.
// ---------------------------------------------------------------------------
__global__ __launch_bounds__(256)
void qkv_proj(const float*  __restrict__ x,
              const bf16_t* __restrict__ WtQ,
              const bf16_t* __restrict__ WtK,
              const bf16_t* __restrict__ WtV,
              const float*  __restrict__ bq,
              const float*  __restrict__ bk,
              const float*  __restrict__ bv,
              bf16_t* __restrict__ qb,
              bf16_t* __restrict__ kb,
              bf16_t* __restrict__ vbT) {
  const int lane = threadIdx.x & 31;
  const int wave = threadIdx.x >> 5;
  const int nl   = lane & 15;
  const int half = lane >> 4;
  const long rowBase = (long)blockIdx.x * 128 + wave * 16;
  const long batch   = rowBase / NKEYS;          // 4096 % 128 == 0: uniform
  const long nBase   = rowBase - batch * NKEYS;  // row within batch

  // A fragments: 16x32 bf16 (ISA 7.12.2): chunks at half*8 and 16+half*8.
  const float* xrow = x + (rowBase + nl) * (long)CDIM;
  BFrag a[8];
#pragma unroll
  for (int ks = 0; ks < 8; ++ks) {
    const float* p0 = xrow + ks * 32 + half * 8;
    const float* p1 = xrow + ks * 32 + 16 + half * 8;
#pragma unroll
    for (int t = 0; t < 8; ++t) {
      a[ks].h[t]     = (bf16_t)p0[t];
      a[ks].h[8 + t] = (bf16_t)p1[t];
    }
  }

#pragma unroll
  for (int ct = 0; ct < 20; ++ct) {
    const bf16_t* Wt; const float* bias; int colLocal;
    if (ct < 2)      { Wt = WtQ; bias = bq; colLocal = ct * 16 + nl; }
    else if (ct < 4) { Wt = WtK; bias = bk; colLocal = (ct - 2) * 16 + nl; }
    else             { Wt = WtV; bias = bv; colLocal = (ct - 4) * 16 + nl; }

    FAcc acc;
#pragma unroll
    for (int r = 0; r < 8; ++r) acc.f[r] = 0.0f;

#pragma unroll
    for (int ks = 0; ks < 8; ++ks) {
      BFrag b;   // B 32x16: K chunks at ks*32 + half*8 and ks*32 + 16 + half*8
      const bf16_t* wp = Wt + (long)colLocal * 256 + ks * 32 + half * 8;
      *(uint4*)&b.h[0] = *(const uint4*)wp;
      *(uint4*)&b.h[8] = *(const uint4*)(wp + 16);
      acc.v = __builtin_amdgcn_wmma_f32_16x16x32_bf16(
          false, a[ks].v, false, b.v, (short)0, acc.v, false, false);
    }

    const float bscal = bias[colLocal];
#pragma unroll
    for (int r = 0; r < 8; ++r) {
      long row = rowBase + r + 8 * half;  // C/D layout: VGPR r -> M = r+8*half
      bf16_t val = (bf16_t)(acc.f[r] + bscal);
      if (ct < 2)      qb[row * DDIM + colLocal] = val;
      else if (ct < 4) kb[row * DDIM + colLocal] = val;
      else {
        long n = nBase + r + 8 * half;
        vbT[(batch * CDIM + colLocal) * (long)NKEYS + n] = val;
      }
    }
  }
}

// ---------------------------------------------------------------------------
// Kernel 3: flash attention.  Block = 8 waves x 16 rows = 128 Q rows of one
// batch.  K ([key][d]) and V^T ([c][key]) tiles are double-buffered in LDS
// via GLOBAL_LOAD_ASYNC_TO_LDS_B128 (ASYNCcnt), overlapping the next tile's
// DMA with the current tile's 18 WMMAs.  Online softmax in fp32 in the C/D
// register layout; P re-laid-out to an A fragment via per-wave LDS scratch.
// ---------------------------------------------------------------------------
__global__ __launch_bounds__(256)
void flash_attn(const float*  __restrict__ x,
                const bf16_t* __restrict__ qb,
                const bf16_t* __restrict__ kb,
                const bf16_t* __restrict__ vbT,
                const float*  __restrict__ gamma_p,
                float* __restrict__ out) {
  __shared__ bf16_t sK[2][32][32];     //  4 KB : K tiles, [key][d]
  __shared__ bf16_t sV[2][CDIM][32];   // 32 KB : V^T tiles, [c][key]
  __shared__ bf16_t sP[8][16][32];     //  8 KB : per-wave P relayout scratch

  const int tid  = threadIdx.x;
  const int lane = tid & 31;
  const int wave = tid >> 5;
  const int nl   = lane & 15;
  const int half = lane >> 4;

  const int  batch    = blockIdx.x >> 5;                      // 32 blocks/batch
  const int  rowBase  = (blockIdx.x & 31) * 128 + wave * 16;  // within batch
  const long batchOff = (long)batch * NKEYS;

  const bf16_t* Kbat  = kb  + batchOff * DDIM;
  const bf16_t* VbatT = vbT + (long)batch * CDIM * NKEYS;

  // Persistent Q fragment (A 16x32 bf16)
  BFrag qf;
  {
    const bf16_t* qrow = qb + (batchOff + rowBase + nl) * (long)DDIM;
    *(uint4*)&qf.h[0] = *(const uint4*)(qrow + half * 8);
    *(uint4*)&qf.h[8] = *(const uint4*)(qrow + 16 + half * 8);
  }

  FAcc acc[16];
#pragma unroll
  for (int t = 0; t < 16; ++t)
#pragma unroll
    for (int r = 0; r < 8; ++r) acc[t].f[r] = 0.0f;

  float mrow[8], lrow[8];
#pragma unroll
  for (int r = 0; r < 8; ++r) { mrow[r] = -3.0e38f; lrow[r] = 0.0f; }

  // ---- tile staging: straight contiguous copies (V already transposed) ----
  auto stage_tile = [&](int buf, int n0) {
    // V^T tile: 256 rows x 32 keys = 16 KB -> 1024 x 16B chunks, 4/thread
#pragma unroll
    for (int q = 0; q < 4; ++q) {
      int e = tid + q * 256;
      int c = e >> 2, p = e & 3;
      const bf16_t* src = VbatT + (long)c * NKEYS + n0 + p * 8;
      bf16_t* dst = &sV[buf][c][p * 8];
#if USE_ASYNC_LDS
      __builtin_amdgcn_global_load_async_to_lds_b128(GAS1(src), LAS3(dst), 0, 0);
#else
      *(uint4*)dst = *(const uint4*)src;
#endif
    }
    // K tile: 64 rows x 16 d = 2 KB -> 128 x 16B chunks, threads 0..127
    if (tid < 128) {
      int n = tid >> 1, p = tid & 1;
      const bf16_t* src = Kbat + (long)(n0 + n) * DDIM + p * 8;
      bf16_t* dst = &sK[buf][n][p * 8];
#if USE_ASYNC_LDS
      __builtin_amdgcn_global_load_async_to_lds_b128(GAS1(src), LAS3(dst), 0, 0);
#else
      *(uint4*)dst = *(const uint4*)src;
#endif
    }
  };

  stage_tile(0, 0);   // prologue: fill buffer 0

#pragma unroll 1
  for (int i = 0; i < NTILES; ++i) {
    const int cur = i & 1;
#if USE_ASYNC_LDS
    __builtin_amdgcn_s_wait_asynccnt(0);  // own async ops done (pre-barrier)
#endif
    __syncthreads();                       // all waves' tile data visible
    if (i + 1 < NTILES) stage_tile(cur ^ 1, (i + 1) * 32);  // overlap DMA

    // ---- S = Q * K^T : two 16x16 tiles ----
    BFrag bk0, bk1;
    *(uint4*)&bk0.h[0] = *(const uint4*)&sK[cur][nl][half * 8];
    *(uint4*)&bk0.h[8] = *(const uint4*)&sK[cur][nl][16 + half * 8];
    *(uint4*)&bk1.h[0] = *(const uint4*)&sK[cur][16 + nl][half * 8];
    *(uint4*)&bk1.h[8] = *(const uint4*)&sK[cur][16 + nl][16 + half * 8];

    FAcc z;
#pragma unroll
    for (int r = 0; r < 8; ++r) z.f[r] = 0.0f;

    FAcc S0, S1;
    S0.v = __builtin_amdgcn_wmma_f32_16x16x32_bf16(
        false, qf.v, false, bk0.v, (short)0, z.v, false, false);
    S1.v = __builtin_amdgcn_wmma_f32_16x16x32_bf16(
        false, qf.v, false, bk1.v, (short)0, z.v, false, false);

    // ---- online softmax (C/D layout: VGPR r = row r+8*half, lane = col) ----
    float tm[8];
#pragma unroll
    for (int r = 0; r < 8; ++r) tm[r] = fmaxf(S0.f[r], S1.f[r]);
#pragma unroll
    for (int off = 1; off < 16; off <<= 1)
#pragma unroll
      for (int r = 0; r < 8; ++r) tm[r] = fmaxf(tm[r], __shfl_xor(tm[r], off, 32));

    float scale[8];
#pragma unroll
    for (int r = 0; r < 8; ++r) {
      float mn = fmaxf(mrow[r], tm[r]);
      scale[r] = __expf(mrow[r] - mn);
      mrow[r]  = mn;
    }
    float ps[8];
#pragma unroll
    for (int r = 0; r < 8; ++r) {
      S0.f[r] = __expf(S0.f[r] - mrow[r]);
      S1.f[r] = __expf(S1.f[r] - mrow[r]);
      ps[r]   = S0.f[r] + S1.f[r];
    }
#pragma unroll
    for (int off = 1; off < 16; off <<= 1)
#pragma unroll
      for (int r = 0; r < 8; ++r) ps[r] += __shfl_xor(ps[r], off, 32);
#pragma unroll
    for (int r = 0; r < 8; ++r) lrow[r] = lrow[r] * scale[r] + ps[r];

#pragma unroll
    for (int t = 0; t < 16; ++t)
#pragma unroll
      for (int r = 0; r < 8; ++r) acc[t].f[r] *= scale[r];

    // ---- P: C/D layout -> A fragment, via per-wave LDS scratch ----
#pragma unroll
    for (int r = 0; r < 8; ++r) {
      sP[wave][r + 8 * half][nl]      = (bf16_t)S0.f[r];
      sP[wave][r + 8 * half][16 + nl] = (bf16_t)S1.f[r];
    }
    BFrag pf;
    {
      const bf16_t* pr = &sP[wave][nl][0];
      *(uint4*)&pf.h[0] = *(const uint4*)(pr + half * 8);
      *(uint4*)&pf.h[8] = *(const uint4*)(pr + 16 + half * 8);
    }

    // ---- acc += P * V : 16 column tiles of 16 ----
#pragma unroll
    for (int t = 0; t < 16; ++t) {
      BFrag bvf;
      *(uint4*)&bvf.h[0] = *(const uint4*)&sV[cur][t * 16 + nl][half * 8];
      *(uint4*)&bvf.h[8] = *(const uint4*)&sV[cur][t * 16 + nl][16 + half * 8];
      acc[t].v = __builtin_amdgcn_wmma_f32_16x16x32_bf16(
          false, pf.v, false, bvf.v, (short)0, acc[t].v, false, false);
    }
  }

  // ---- epilogue: out = gamma * (acc / l) + x ----
  const float g = *gamma_p;
  float linv[8];
#pragma unroll
  for (int r = 0; r < 8; ++r) linv[r] = 1.0f / lrow[r];

#pragma unroll
  for (int t = 0; t < 16; ++t) {
#pragma unroll
    for (int r = 0; r < 8; ++r) {
      long row = batchOff + rowBase + r + 8 * half;
      long idx = row * CDIM + t * 16 + nl;
      out[idx] = g * acc[t].f[r] * linv[r] + x[idx];
    }
  }
}

// ---------------------------------------------------------------------------
// Launch.  ws layout (bf16 elements):
//   WtQ 32*256 | WtK 32*256 | WtV 256*256 | q 16384*32 | k 16384*32
//   | vT 4*256*4096   -> total 10,649,600 bytes
// ---------------------------------------------------------------------------
extern "C" void kernel_launch(void* const* d_in, const int* in_sizes, int n_in,
                              void* d_out, int out_size, void* d_ws, size_t ws_size,
                              hipStream_t stream) {
  (void)in_sizes; (void)n_in; (void)out_size; (void)ws_size;

  const float* x     = (const float*)d_in[0];
  const float* Wq    = (const float*)d_in[1];
  const float* bq    = (const float*)d_in[2];
  const float* Wk    = (const float*)d_in[3];
  const float* bk    = (const float*)d_in[4];
  const float* Wv    = (const float*)d_in[5];
  const float* bv    = (const float*)d_in[6];
  const float* gamma = (const float*)d_in[7];
  float* out = (float*)d_out;

  bf16_t* WtQ  = (bf16_t*)d_ws;
  bf16_t* WtK  = WtQ  + 32 * 256;
  bf16_t* WtV  = WtK  + 32 * 256;
  bf16_t* qbuf = WtV  + 256 * 256;
  bf16_t* kbuf = qbuf + (long)NROWS * DDIM;
  bf16_t* vbufT = kbuf + (long)NROWS * DDIM;   // [batch][c][n]

  prep_weights<<<320, 256, 0, stream>>>(Wq, Wk, Wv, WtQ, WtK, WtV);
  qkv_proj<<<NROWS / 128, 256, 0, stream>>>(x, WtQ, WtK, WtV, bq, bk, bv,
                                            qbuf, kbuf, vbufT);
  flash_attn<<<128, 256, 0, stream>>>(x, qbuf, kbuf, vbufT, gamma, out);
}
